// CPLSTM_14972255994405
// MI455X (gfx1250) — matrix operand order repaired and checked by reading, not compiled, not run on hardware
//
#include <hip/hip_runtime.h>
#include <hip/hip_bf16.h>

// ---------------------------------------------------------------------------
// CP-LSTM on gfx1250 (MI455X).  bf16 WMMA (16x16x32, f32 accum).
// All matrices held in pre-swizzled per-lane WMMA fragment layouts
// (ISA 7.12.2); the recurrent operands g/h are produced *in* fragment layout
// so the scan's inner loops are pure {32B frag load -> v_wmma}.
// ---------------------------------------------------------------------------

typedef __attribute__((ext_vector_type(16))) __bf16 v16bf;
typedef __attribute__((ext_vector_type(8)))  float  v8f;

#define INPUT  1024
#define HIDDEN 1024
#define R4     1024          // RANK*4
#define BATCH  32
#define SEQ    512
#define MS     (BATCH * SEQ) // 16384 rows for the xb GEMM
#define KT32   32            // 1024 / 32 K-tiles

// float -> bf16, round-to-nearest-even
__device__ __forceinline__ __bf16 f2bf(float f) {
    unsigned u = __builtin_bit_cast(unsigned, f);
    u += 0x7FFFu + ((u >> 16) & 1u);
    unsigned short s = (unsigned short)(u >> 16);
    return __builtin_bit_cast(__bf16, s);
}

__device__ __forceinline__ v8f wmma_bf16(v16bf a, v16bf b, v8f c) {
    return __builtin_amdgcn_wmma_f32_16x16x32_bf16(false, a, false, b,
                                                   (short)0, c, false, false);
}

__device__ __forceinline__ float sigmoidf_(float x) {
    return 1.0f / (1.0f + __expf(-x));
}

// ---------------------------------------------------------------------------
// Prep: swizzle fp32 row-major (rows x cols) into bf16 A-fragment storage:
//   elems 0..7  -> K = kt*32 + (lane>>4)*8 + j      (lane&15 = row)
//   elems 8..15 -> K = kt*32 + 16 + (lane>>4)*8 + j
//   dst[((mt*KT + kt)*32 + lane)*16 + j]
// ---------------------------------------------------------------------------
__global__ void swizzleA_kernel(const float* __restrict__ src, __bf16* __restrict__ dst,
                                int rows, int cols) {
    long long idx = (long long)blockIdx.x * blockDim.x + threadIdx.x;
    if (idx >= (long long)rows * cols) return;
    int m = (int)(idx / cols), k = (int)(idx % cols);
    int mt = m >> 4, mrow = m & 15;
    int kt = k >> 5, krem = k & 31;
    int khalf = (krem & 15) >> 3;
    int j = (krem >= 16 ? 8 : 0) + (krem & 7);
    int lane = khalf * 16 + mrow;
    int KT = cols >> 5;
    dst[(((size_t)mt * KT + kt) * 32 + lane) * 16 + j] = f2bf(src[idx]);
}

// ---------------------------------------------------------------------------
// Prep: swizzle fp32 row-major (K x N) into bf16 B-fragment storage:
//   lane&15 = col, elems 0..7 -> K = kt*32 + (lane>>4)*8 + j, elems 8..15 -> K = kt*32 + 16 + (lane>>4)*8 + j
//   dst[((nt*KT + kt)*32 + lane)*16 + j]
// ---------------------------------------------------------------------------
__global__ void swizzleB_kernel(const float* __restrict__ src, __bf16* __restrict__ dst,
                                int K, int N) {
    long long idx = (long long)blockIdx.x * blockDim.x + threadIdx.x;
    if (idx >= (long long)K * N) return;
    int k = (int)(idx / N), n = (int)(idx % N);
    int kt = k >> 5, krem = k & 31;
    int khalf = (krem & 15) >> 3, j = (krem >= 16 ? 8 : 0) + (krem & 7);
    int lane = khalf * 16 + (n & 15);
    int nt = n >> 4;
    int KT = K >> 5;
    dst[(((size_t)nt * KT + kt) * 32 + lane) * 16 + j] = f2bf(src[idx]);
}

__global__ void zero_u32_kernel(unsigned* __restrict__ p, int n) {
    int i = blockIdx.x * blockDim.x + threadIdx.x;
    if (i < n) p[i] = 0u;
}

// ---------------------------------------------------------------------------
// Phase 1: xb = x @ b  (16384x1024)@(1024x1024) -> stored TRANSPOSED as
// xbT(SEQ, BATCH, 4R) so the scan reads one contiguous slab per step.
// Wave computes 16x64 (4 accs); K-loop software-pipelined one deep.
// grid: 128 m-blocks * 16 n-blocks = 2048, block = 256 (8 waves)
// ---------------------------------------------------------------------------
__global__ void xb_gemm_kernel(const __bf16* __restrict__ xA,
                               const __bf16* __restrict__ bB,
                               float* __restrict__ xbT) {
    int w = threadIdx.x >> 5, lane = threadIdx.x & 31;
    int mblk = blockIdx.x >> 4;
    int nblk = blockIdx.x & 15;
    int mt = mblk * 8 + w;         // 0..1023
    int ntb = nblk * 4;            // 0..60

    const __bf16* ap = xA + ((size_t)mt * KT32) * 512 + lane * 16;
    const __bf16* bp = bB + ((size_t)ntb * KT32) * 512 + lane * 16;
    const size_t nstep = (size_t)KT32 * 512;   // one N-tile of B fragments

    v8f acc0 = {}, acc1 = {}, acc2 = {}, acc3 = {};
    v16bf af = *(const v16bf*)ap;
    v16bf b0 = *(const v16bf*)(bp);
    v16bf b1 = *(const v16bf*)(bp + nstep);
    v16bf b2 = *(const v16bf*)(bp + 2 * nstep);
    v16bf b3 = *(const v16bf*)(bp + 3 * nstep);
#pragma unroll 2
    for (int kt = 0; kt < KT32 - 1; ++kt) {
        ap += 512; bp += 512;
        v16bf afn = *(const v16bf*)ap;
        v16bf b0n = *(const v16bf*)(bp);
        v16bf b1n = *(const v16bf*)(bp + nstep);
        v16bf b2n = *(const v16bf*)(bp + 2 * nstep);
        v16bf b3n = *(const v16bf*)(bp + 3 * nstep);
        acc0 = wmma_bf16(af, b0, acc0);
        acc1 = wmma_bf16(af, b1, acc1);
        acc2 = wmma_bf16(af, b2, acc2);
        acc3 = wmma_bf16(af, b3, acc3);
        af = afn; b0 = b0n; b1 = b1n; b2 = b2n; b3 = b3n;
    }
    acc0 = wmma_bf16(af, b0, acc0);
    acc1 = wmma_bf16(af, b1, acc1);
    acc2 = wmma_bf16(af, b2, acc2);
    acc3 = wmma_bf16(af, b3, acc3);

    // C/D layout: lane&15 = col, (lane>>4)*8 + r = row
    int col = lane & 15;
    int rowb = mt * 16 + ((lane >> 4) << 3);
#pragma unroll
    for (int r = 0; r < 8; ++r) {
        int m = rowb + r;              // global row = b*SEQ + s
        int bb = m >> 9;               // / SEQ
        int ss = m & (SEQ - 1);
        float* o = xbT + ((size_t)ss * BATCH + bb) * R4 + ntb * 16 + col;
        o[0] = acc0[r]; o[16] = acc1[r]; o[32] = acc2[r]; o[48] = acc3[r];
    }
}

// ---------------------------------------------------------------------------
// Scan stage A (step t): g = (h @ a) .* xb_t.  hA is bf16 in A-fragment
// layout (written by stage B); result g written straight into A-fragment
// layout for stage B.  One 16x16 tile per wave; grid 16 x 256.
// ---------------------------------------------------------------------------
__global__ void lstm_stageA_kernel(const __bf16* __restrict__ hA,
                                   const __bf16* __restrict__ aB,
                                   const float* __restrict__ xbT,
                                   __bf16* __restrict__ gA, int t) {
    int w = threadIdx.x >> 5, lane = threadIdx.x & 31;
    int gw = blockIdx.x * 8 + w;       // 0..127
    int mt = gw >> 6;                  // 0..1
    int nt = gw & 63;                  // 0..63

    const __bf16* ap = hA + ((size_t)mt * KT32) * 512 + lane * 16;
    const __bf16* bp = aB + ((size_t)nt * KT32) * 512 + lane * 16;

    v8f acc = {};
    v16bf af = *(const v16bf*)ap;
    v16bf bf = *(const v16bf*)bp;
#pragma unroll 4
    for (int kt = 0; kt < KT32 - 1; ++kt) {
        ap += 512; bp += 512;
        v16bf afn = *(const v16bf*)ap;
        v16bf bfn = *(const v16bf*)bp;
        acc = wmma_bf16(af, bf, acc);
        af = afn; bf = bfn;
    }
    acc = wmma_bf16(af, bf, acc);

    int col = nt * 16 + (lane & 15);
    int rowb = mt * 16 + ((lane >> 4) << 3);
    const float* xrow = xbT + (size_t)t * BATCH * R4;

    // scatter into gA A-fragment layout (k = col fixed per lane)
    int kt2 = col >> 5, krem = col & 31;
    int khalf = (krem & 15) >> 3;
    int j = (krem >= 16 ? 8 : 0) + (krem & 7);
    __bf16* gbase = gA + (((size_t)mt * KT32 + kt2) * 32 + khalf * 16) * 16 + j;
#pragma unroll
    for (int r = 0; r < 8; ++r) {
        int m = rowb + r;
        float v = acc[r] * xrow[(size_t)m * R4 + col];
        gbase[(size_t)(m & 15) * 16] = f2bf(v);
    }
}

// ---------------------------------------------------------------------------
// Scan stage B (step t): 4 gate projections + fused LSTM cell update.
// gA is bf16 A-fragment layout; gate k owns K-tiles [8k, 8k+8).
// Writes h back into hA fragment layout, hidden_seq (f32), and at the last
// step also the h_t / c_t output tails.  Wave = one 16x16 H-tile, 4 accs.
// ---------------------------------------------------------------------------
__global__ void lstm_stageB_kernel(const __bf16* __restrict__ gA,
                                   const __bf16* __restrict__ ctB,
                                   __bf16* __restrict__ hA,
                                   float* __restrict__ c,
                                   float* __restrict__ out, int t) {
    int w = threadIdx.x >> 5, lane = threadIdx.x & 31;
    int gw = blockIdx.x * 8 + w;       // 0..127
    int mt = gw >> 6;
    int nt = gw & 63;

    const __bf16* ap = gA + ((size_t)mt * KT32) * 512 + lane * 16;
    const __bf16* bp = ctB + ((size_t)nt * KT32) * 512 + lane * 16;

    v8f acc[4] = {v8f{}, v8f{}, v8f{}, v8f{}};
    v16bf af = *(const v16bf*)ap;
    v16bf bf = *(const v16bf*)bp;
#pragma unroll
    for (int kt = 0; kt < KT32 - 1; ++kt) {
        ap += 512; bp += 512;
        v16bf afn = *(const v16bf*)ap;
        v16bf bfn = *(const v16bf*)bp;
        acc[kt >> 3] = wmma_bf16(af, bf, acc[kt >> 3]);
        af = afn; bf = bfn;
    }
    acc[3] = wmma_bf16(af, bf, acc[3]);

    int col = nt * 16 + (lane & 15);
    int rowb = mt * 16 + ((lane >> 4) << 3);

    // hA scatter params (k = col)
    int kt2 = col >> 5, krem = col & 31;
    int khalf = (krem & 15) >> 3;
    int j = (krem >= 16 ? 8 : 0) + (krem & 7);
    __bf16* hbase = hA + (((size_t)mt * KT32 + kt2) * 32 + khalf * 16) * 16 + j;

    bool last = (t == SEQ - 1);
#pragma unroll
    for (int r = 0; r < 8; ++r) {
        int m = rowb + r;
        float fg = sigmoidf_(acc[0][r]);
        float ig = sigmoidf_(acc[1][r]);
        float gg = tanhf(acc[2][r]);
        float og = sigmoidf_(acc[3][r]);
        size_t idx = (size_t)m * HIDDEN + col;
        float cv = fg * c[idx] + ig * gg;
        float hv = og * tanhf(cv);
        c[idx] = cv;
        hbase[(size_t)(m & 15) * 16] = f2bf(hv);
        out[((size_t)m * SEQ + t) * HIDDEN + col] = hv;
        if (last) {
            out[(size_t)MS * HIDDEN + idx] = hv;                           // h_t
            out[(size_t)MS * HIDDEN + (size_t)BATCH * HIDDEN + idx] = cv;  // c_t
        }
    }
}

// ---------------------------------------------------------------------------
extern "C" void kernel_launch(void* const* d_in, const int* in_sizes, int n_in,
                              void* d_out, int out_size, void* d_ws, size_t ws_size,
                              hipStream_t stream) {
    const float* x  = (const float*)d_in[0];   // (32, 512, 1024)
    const float* a  = (const float*)d_in[1];   // (1024, 1024)
    const float* b  = (const float*)d_in[2];   // (1024, 1024)
    const float* ct = (const float*)d_in[3];   // (1024, 1024)
    float* out = (float*)d_out;                // hidden_seq | h_t | c_t

    char* ws = (char*)d_ws;
    size_t off = 0;
    __bf16* xA  = (__bf16*)(ws + off); off += (size_t)MS * INPUT * 2;      // 32 MB
    __bf16* bB  = (__bf16*)(ws + off); off += (size_t)INPUT * R4 * 2;      // 2 MB
    __bf16* aB  = (__bf16*)(ws + off); off += (size_t)HIDDEN * R4 * 2;     // 2 MB
    __bf16* ctB = (__bf16*)(ws + off); off += (size_t)R4 * HIDDEN * 2;     // 2 MB
    float*  xbT = (float*)(ws + off);  off += (size_t)SEQ * BATCH * R4 * 4;// 64 MB
    float*  c   = (float*)(ws + off);  off += (size_t)BATCH * HIDDEN * 4;  // 128 KB
    __bf16* hA  = (__bf16*)(ws + off); off += (size_t)BATCH * HIDDEN * 2;  // 64 KB
    __bf16* gA  = (__bf16*)(ws + off); off += (size_t)BATCH * R4 * 2;      // 64 KB

    // --- prep: swizzle weights/x, zero carried state (c fp32 + hA bf16) ---
    {
        long long n = (long long)MS * INPUT;
        swizzleA_kernel<<<(unsigned)((n + 255) / 256), 256, 0, stream>>>(x, xA, MS, INPUT);
    }
    swizzleB_kernel<<<(INPUT * R4) / 256, 256, 0, stream>>>(b, bB, INPUT, R4);
    swizzleB_kernel<<<(HIDDEN * R4) / 256, 256, 0, stream>>>(a, aB, HIDDEN, R4);
    swizzleB_kernel<<<(R4 * HIDDEN) / 256, 256, 0, stream>>>(ct, ctB, R4, HIDDEN);
    {
        // c (128KB) and hA (64KB) are adjacent: zero 49152 dwords in one shot
        int nwords = (BATCH * HIDDEN * 4 + BATCH * HIDDEN * 2) / 4;
        zero_u32_kernel<<<(nwords + 255) / 256, 256, 0, stream>>>((unsigned*)c, nwords);
    }

    // --- phase 1: xbT = (x @ b) transposed to (SEQ, BATCH, 4R) ---
    xb_gemm_kernel<<<2048, 256, 0, stream>>>(xA, bB, xbT);

    // --- phase 2: sequential scan ---
    for (int t = 0; t < SEQ; ++t) {
        lstm_stageA_kernel<<<16, 256, 0, stream>>>(hA, aB, xbT, gA, t);
        lstm_stageB_kernel<<<16, 256, 0, stream>>>(gA, ctB, hA, c, out, t);
    }
}
